// GCN_25383256719664
// MI455X (gfx1250) — matrix operand order repaired and checked by reading, not compiled, not run on hardware
//
#include <hip/hip_runtime.h>

#define NN 30000
#define NE 480000
#define NG 500

typedef __bf16 bf16_t;
typedef bf16_t v16bf __attribute__((ext_vector_type(16)));
typedef float  v8f   __attribute__((ext_vector_type(8)));

__device__ __forceinline__ unsigned short f2bf(float f) {
  unsigned int u = __float_as_uint(f);
  u += 0x7FFFu + ((u >> 16) & 1u);   // round-to-nearest-even
  return (unsigned short)(u >> 16);
}

// ------------------------------------------------------------------
// WMMA GEMM: C[M,N] = A[M,Kp] @ B[Kp,N] (+bias)(+relu)
// A: bf16 bits, row-major, leading dim lda (multiple of 32)
// Bt: bf16 bits, B transposed, [Npad rows][Kp], ldb = Kp
// One 16x16 tile per wave, 4 waves (4 M-tiles) per 128-thread block.
// Fragment layouts per CDNA5 ISA 7.12.2:
//   A (16x32 bf16): lane half h, row m=lane&15; K runs {k0+8h..+7} -> elems 0..7,
//                   {k0+8h+16..+23} -> elems 8..15  (two 16B loads)
//   B (32x16 bf16): lane half h, col n=lane&15; K runs {k0+8h..+7} -> elems 0..7,
//                   {k0+8h+16..+23} -> elems 8..15  (same as A)
//   C/D f32: elem e -> m = m0 + 8h + e, n = n0 + (lane&15)
// ------------------------------------------------------------------
__global__ __launch_bounds__(128) void k_wmma_gemm(
    const unsigned short* __restrict__ A, int lda,
    const unsigned short* __restrict__ Bt, int ldb,
    const float* __restrict__ bias,
    float* __restrict__ Cf, int ldc,
    unsigned short* __restrict__ Cb, int ldcb,
    int M, int N, int Kp, int relu)
{
  const int lane = threadIdx.x & 31;
  const int wid  = threadIdx.x >> 5;
  const int hl   = lane >> 4;
  const int lid  = lane & 15;
  const int n0 = blockIdx.x * 16;
  const int m0 = (blockIdx.y * 4 + wid) * 16;

  int arow = m0 + lid; if (arow > M - 1) arow = M - 1;   // clamp (stores guarded)
  const unsigned short* aptr = A  + (size_t)arow * lda + hl * 8;
  const unsigned short* bptr = Bt + (size_t)(n0 + lid) * ldb + hl * 8; // Bt padded to Npad rows

  v8f acc = {0.f, 0.f, 0.f, 0.f, 0.f, 0.f, 0.f, 0.f};
  for (int k0 = 0; k0 < Kp; k0 += 32) {
    union { v16bf v; uint4 q[2]; } a, b;
    a.q[0] = *(const uint4*)(aptr + k0);
    a.q[1] = *(const uint4*)(aptr + k0 + 16);
    b.q[0] = *(const uint4*)(bptr + k0);
    b.q[1] = *(const uint4*)(bptr + k0 + 16);
    acc = __builtin_amdgcn_wmma_f32_16x16x32_bf16(false, a.v, false, b.v,
                                                  (short)0, acc, false, false);
  }

  const int n = n0 + lid;
  if (n < N) {
    const float bv = bias ? bias[n] : 0.f;
#pragma unroll
    for (int e = 0; e < 8; ++e) {
      int m = m0 + hl * 8 + e;
      if (m < M) {
        float v = acc[e] + bv;
        if (relu) v = fmaxf(v, 0.f);
        if (Cf) Cf[(size_t)m * ldc + n] = v;
        if (Cb) Cb[(size_t)m * ldcb + n] = f2bf(v);
      }
    }
  }
}

// ------------------------- helper kernels -------------------------
__global__ void k_zero_u32(unsigned int* p, int n) {
  int i = blockIdx.x * blockDim.x + threadIdx.x;
  if (i < n) p[i] = 0u;
}

__global__ void k_deg(const int* __restrict__ dst, int* __restrict__ deg, int nE) {
  int i = blockIdx.x * blockDim.x + threadIdx.x;
  if (i < nE) atomicAdd(&deg[dst[i]], 1);
}

__global__ void k_dinv(const int* __restrict__ deg, float* __restrict__ dinv, int n) {
  int i = blockIdx.x * blockDim.x + threadIdx.x;
  if (i < n) dinv[i] = rsqrtf((float)deg[i] + 1.0f);
}

// f32 [rows,din] -> bf16 [rows,Kp] zero-padded
__global__ void k_cvt_pad(const float* __restrict__ x, unsigned short* __restrict__ xb,
                          int din, int Kp, int total) {
  int idx = blockIdx.x * blockDim.x + threadIdx.x;
  if (idx >= total) return;
  int i = idx / Kp, f = idx - i * Kp;
  xb[idx] = (f < din) ? f2bf(x[(size_t)i * din + f]) : (unsigned short)0;
}

// W [din,dout] f32 -> Wt [Npad,Kp] bf16, Wt[n*Kp+k] = W[k*dout+n]
__global__ void k_w2bt(const float* __restrict__ w, unsigned short* __restrict__ wt,
                       int din, int dout, int Kp, int total) {
  int idx = blockIdx.x * blockDim.x + threadIdx.x;
  if (idx >= total) return;
  int n = idx / Kp, k = idx - n * Kp;
  wt[idx] = (k < din && n < dout) ? f2bf(w[(size_t)k * dout + n]) : (unsigned short)0;
}

// self-loop: agg = xw * dinv^2
__global__ void k_selfinit(const float* __restrict__ xw, const float* __restrict__ dinv,
                           float* __restrict__ agg, int dout, int total) {
  int idx = blockIdx.x * blockDim.x + threadIdx.x;
  if (idx >= total) return;
  int i = idx / dout;
  float di = dinv[i];
  agg[idx] = xw[idx] * di * di;
}

// edge scatter: agg[dst] += xw[src] * dinv[src]*dinv[dst]   (one block per edge)
__global__ __launch_bounds__(128) void k_edge(const int* __restrict__ src, const int* __restrict__ dst,
                                              const float* __restrict__ dinv,
                                              const float* __restrict__ xw,
                                              float* __restrict__ agg, int dout) {
  int e = blockIdx.x;
  int s = src[e], d = dst[e];
  float w = dinv[s] * dinv[d];
  const float* xs = xw + (size_t)s * dout;
  float* ad = agg + (size_t)d * dout;
  for (int f = threadIdx.x; f < dout; f += blockDim.x)
    atomicAdd(&ad[f], xs[f] * w);
}

// h = relu(agg + b); write f32 h (reuses xw buffer) and bf16 padded next-input
__global__ void k_post(const float* __restrict__ agg, const float* __restrict__ b,
                       float* __restrict__ h, unsigned short* __restrict__ hb,
                       int dout, int Kp, int total) {
  int idx = blockIdx.x * blockDim.x + threadIdx.x;
  if (idx >= total) return;
  int i = idx / Kp, f = idx - i * Kp;
  if (f < dout) {
    float v = fmaxf(agg[(size_t)i * dout + f] + b[f], 0.0f);
    h[(size_t)i * dout + f] = v;
    hb[idx] = f2bf(v);
  } else {
    hb[idx] = 0;
  }
}

// segment max via uint-bit atomicMax (valid: h >= 0 post-relu; init 0 == empty-segment guard)
__global__ void k_pool(const float* __restrict__ h, const int* __restrict__ batch,
                       unsigned int* __restrict__ pooled, int dfeat, int total) {
  int idx = blockIdx.x * blockDim.x + threadIdx.x;
  if (idx >= total) return;
  int i = idx / dfeat, f = idx - i * dfeat;
  atomicMax(&pooled[(size_t)batch[i] * dfeat + f], __float_as_uint(h[idx]));
}

__global__ void k_pool2bf(const unsigned int* __restrict__ pooled, unsigned short* __restrict__ pbf,
                          int dfeat, int Kp, int total) {
  int idx = blockIdx.x * blockDim.x + threadIdx.x;
  if (idx >= total) return;
  int g = idx / Kp, f = idx - g * Kp;
  pbf[idx] = (f < dfeat) ? f2bf(__uint_as_float(pooled[(size_t)g * dfeat + f])) : (unsigned short)0;
}

// ------------------------------------------------------------------
extern "C" void kernel_launch(void* const* d_in, const int* in_sizes, int n_in,
                              void* d_out, int out_size, void* d_ws, size_t ws_size,
                              hipStream_t stream) {
  const float* x     = (const float*)d_in[0];
  const int*   ei    = (const int*)d_in[1];
  const int*   batch = (const int*)d_in[2];
  const float* W1 = (const float*)d_in[3];  const float* b1 = (const float*)d_in[4];
  const float* W2 = (const float*)d_in[5];  const float* b2 = (const float*)d_in[6];
  const float* W3 = (const float*)d_in[7];  const float* b3 = (const float*)d_in[8];
  const float* Wf1 = (const float*)d_in[9]; const float* bf1 = (const float*)d_in[10];
  const float* Wf2 = (const float*)d_in[11]; const float* bf2 = (const float*)d_in[12];
  const int* src = ei;
  const int* dst = ei + NE;
  float* out = (float*)d_out;
  (void)in_sizes; (void)n_in; (void)out_size; (void)ws_size;

  char* base = (char*)d_ws;
  size_t off = 0;
  auto take = [&](size_t bytes) -> char* {
    char* p = base + off;
    off = (off + bytes + 255) & ~(size_t)255;
    return p;
  };
  float*          dinv   = (float*)take((size_t)NN * 4);
  int*            degi   = (int*)take((size_t)NN * 4);
  unsigned short* Xb     = (unsigned short*)take((size_t)NN * 320 * 2);  // bf16 layer input (ping)
  float*          xw     = (float*)take((size_t)NN * 312 * 4);           // XW, later reused as h
  float*          agg    = (float*)take((size_t)NN * 312 * 4);           // aggregation accumulator
  unsigned short* Wt     = (unsigned short*)take((size_t)1024 * 320 * 2);// transposed weight (reused)
  unsigned int*   pooled = (unsigned int*)take((size_t)NG * 312 * 4);
  unsigned short* pbf    = (unsigned short*)take((size_t)NG * 320 * 2);
  unsigned short* f1b    = (unsigned short*)take((size_t)NG * 1024 * 2);

  const int T = 256;

  // degrees -> dinv
  k_zero_u32<<<(NN + T - 1) / T, T, 0, stream>>>((unsigned int*)degi, NN);
  k_deg<<<(NE + T - 1) / T, T, 0, stream>>>(dst, degi, NE);
  k_dinv<<<(NN + T - 1) / T, T, 0, stream>>>(degi, dinv, NN);

  // x -> bf16 padded to Kp=96
  k_cvt_pad<<<(NN * 96 + T - 1) / T, T, 0, stream>>>(x, Xb, 78, 96, NN * 96);

  struct Layer { const float* W; const float* b; int din, dout, Kp, Npad, KpNext; };
  const Layer layers[3] = {
    { W1, b1,  78,  78,  96,  80,  96 },
    { W2, b2,  78, 156,  96, 160, 160 },
    { W3, b3, 156, 312, 160, 320, 320 },
  };

  for (int li = 0; li < 3; ++li) {
    const Layer L = layers[li];
    // weight -> transposed bf16 [Npad, Kp]
    int wtot = L.Npad * L.Kp;
    k_w2bt<<<(wtot + T - 1) / T, T, 0, stream>>>(L.W, Wt, L.din, L.dout, L.Kp, wtot);
    // xw = Xb @ W   (WMMA)
    dim3 g((L.dout + 15) / 16, (NN + 63) / 64);
    k_wmma_gemm<<<g, 128, 0, stream>>>(Xb, L.Kp, Wt, L.Kp, nullptr,
                                       xw, L.dout, nullptr, 0,
                                       NN, L.dout, L.Kp, 0);
    // self-loop init + edge scatter (L2-resident: 37MB << 192MB L2)
    int ntot = NN * L.dout;
    k_selfinit<<<(ntot + T - 1) / T, T, 0, stream>>>(xw, dinv, agg, L.dout, ntot);
    k_edge<<<NE, 128, 0, stream>>>(src, dst, dinv, xw, agg, L.dout);
    // bias + relu; write f32 h into xw buffer, bf16 next-input into Xb
    int ptot = NN * L.KpNext;
    k_post<<<(ptot + T - 1) / T, T, 0, stream>>>(agg, L.b, xw, Xb, L.dout, L.KpNext, ptot);
  }

  // global max pool (h lives in xw buffer, [NN,312])
  k_zero_u32<<<(NG * 312 + T - 1) / T, T, 0, stream>>>(pooled, NG * 312);
  k_pool<<<(NN * 312 + T - 1) / T, T, 0, stream>>>(xw, batch, pooled, 312, NN * 312);
  k_pool2bf<<<(NG * 320 + T - 1) / T, T, 0, stream>>>(pooled, pbf, 312, 320, NG * 320);

  // FC1: [500,312]@[312,1024] + bf1, relu -> bf16
  {
    int wtot = 1024 * 320;
    k_w2bt<<<(wtot + T - 1) / T, T, 0, stream>>>(Wf1, Wt, 312, 1024, 320, wtot);
    dim3 g(1024 / 16, (NG + 63) / 64);
    k_wmma_gemm<<<g, 128, 0, stream>>>(pbf, 320, Wt, 320, bf1,
                                       nullptr, 0, f1b, 1024,
                                       NG, 1024, 320, 1);
  }
  // FC2: [500,1024]@[1024,128] + bf2 -> f32 out
  {
    int wtot = 128 * 1024;
    k_w2bt<<<(wtot + T - 1) / T, T, 0, stream>>>(Wf2, Wt, 1024, 128, 1024, wtot);
    dim3 g(128 / 16, (NG + 63) / 64);
    k_wmma_gemm<<<g, 128, 0, stream>>>(f1b, 1024, Wt, 1024, bf2,
                                       out, 128, nullptr, 0,
                                       NG, 128, 1024, 0);
  }
}
